// SpectralGNNEncoder_2680059592979
// MI455X (gfx1250) — matrix operand order repaired and checked by reading, not compiled, not run on hardware
//
#include <hip/hip_runtime.h>
#include <hip/hip_bf16.h>
#include <stdint.h>

#define NN  50000
#define HID 128
#define LAT 64
#define NE  800000

typedef __attribute__((ext_vector_type(16))) __bf16 v16bf;
typedef __attribute__((ext_vector_type(8)))  float  v8f;

union Frag32B { uint4 u[2]; v16bf bf; };

static __device__ __forceinline__ unsigned short f32_to_bf16_bits(float f) {
  unsigned int u = __float_as_uint(f);
  unsigned int r = (u + 0x7FFFu + ((u >> 16) & 1u)) >> 16;  // round-to-nearest-even
  return (unsigned short)r;
}
static __device__ __forceinline__ __bf16 f32_to_bf16(float f) {
  unsigned short s = f32_to_bf16_bits(f);
  return __builtin_bit_cast(__bf16, s);
}

// ---------------- degree / normalization ----------------
__global__ __launch_bounds__(256) void k_fill(float* __restrict__ p, float v, int n) {
  int i = blockIdx.x * 256 + threadIdx.x;
  if (i < n) p[i] = v;
}

__global__ __launch_bounds__(256) void k_deg_accum(const int* __restrict__ dst,
                                                   const float* __restrict__ ew,
                                                   float* __restrict__ deg) {
  int e = blockIdx.x * 256 + threadIdx.x;
  if (e < NE)
    __hip_atomic_fetch_add(&deg[dst[e]], ew[e], __ATOMIC_RELAXED, __HIP_MEMORY_SCOPE_AGENT);
}

__global__ __launch_bounds__(256) void k_dinv(const float* __restrict__ deg,
                                              float* __restrict__ dinv, int n) {
  int i = blockIdx.x * 256 + threadIdx.x;
  if (i < n) {
    float d = deg[i];
    dinv[i] = (d > 0.0f) ? rsqrtf(d) : 0.0f;
  }
}

// ---------------- weight repack: fp32 row-major -> bf16 WMMA B-fragment layout ----------------
// Wp index = ((kt*ntiles + nt)*32 + lane)*8 + v ; value packs B[k0][n] (lo) | B[k0+1][n] (hi)
// with n = nt*16 + (lane&15), k0 = kt*32 + (lane>>4)*16 + 2v.
__global__ __launch_bounds__(256) void k_pack_w(const float* __restrict__ W,
                                                unsigned int* __restrict__ Wp,
                                                int K, int Ncols) {
  int idx = blockIdx.x * 256 + threadIdx.x;
  int total = (K * Ncols) >> 1;
  if (idx >= total) return;
  int v    = idx & 7;
  int lane = (idx >> 3) & 31;
  int frag = idx >> 8;
  int ntiles = Ncols >> 4;
  int kt = frag / ntiles;
  int nt = frag - kt * ntiles;
  int n  = (nt << 4) + (lane & 15);
  int k0 = (kt << 5) + ((lane >> 4) << 4) + (v << 1);
  unsigned int lo = f32_to_bf16_bits(W[(size_t)k0 * Ncols + n]);
  unsigned int hi = f32_to_bf16_bits(W[(size_t)(k0 + 1) * Ncols + n]);
  Wp[idx] = lo | (hi << 16);
}

__global__ __launch_bounds__(256) void k_cvt_bf16(const float* __restrict__ in,
                                                  __bf16* __restrict__ out, int n) {
  int i = blockIdx.x * 256 + threadIdx.x;
  if (i < n) out[i] = f32_to_bf16(in[i]);
}

__global__ __launch_bounds__(256) void k_zero4(float4* __restrict__ p, int n4) {
  int i = blockIdx.x * 256 + threadIdx.x;
  if (i < n4) p[i] = make_float4(0.f, 0.f, 0.f, 0.f);
}

// ---------------- bf16 WMMA GEMM: Out[M x Ncols] = A[M x 128] * B + bias ----------------
// One wave per 16x64 strip; K=128 -> 4 unrolled k-steps, 4 WMMAs each.
__global__ __launch_bounds__(128) void k_gemm_bf16(const __bf16* __restrict__ A,
                                                   const unsigned int* __restrict__ Bp,
                                                   const float* __restrict__ bias,
                                                   float* __restrict__ Out,
                                                   int M, int Ncols) {
  const int lane = threadIdx.x & 31;
  const int wid  = (blockIdx.x << 2) + (threadIdx.x >> 5);
  const int nstrips = Ncols >> 6;          // 64-wide strips
  const int mt = wid / nstrips;
  const int ns = wid - mt * nstrips;
  if (mt >= (M >> 4)) return;              // wave-uniform guard (EXEC stays all-1 for WMMA)

  const int m0    = (mt << 4) + (lane & 15);
  const int khalf = lane >> 4;
  const int ntilesTot = Ncols >> 4;
  const int nt0   = ns << 2;

  v8f acc[4] = {};
#pragma unroll
  for (int kt = 0; kt < (HID >> 5); ++kt) {
    Frag32B a;
    const __bf16* ap = A + (size_t)m0 * HID + (kt << 5) + (khalf << 4);
    a.u[0] = *(const uint4*)(ap);
    a.u[1] = *(const uint4*)(ap + 8);
#pragma unroll
    for (int j = 0; j < 4; ++j) {
      Frag32B b;
      const unsigned int* bp = Bp + (((size_t)(kt * ntilesTot + nt0 + j) << 5) + lane) * 8;
      b.u[0] = *(const uint4*)(bp);
      b.u[1] = *(const uint4*)(bp + 4);
      acc[j] = __builtin_amdgcn_wmma_f32_16x16x32_bf16(
          /*neg_a=*/false, a.bf, /*neg_b=*/false, b.bf,
          /*c_mod=*/(short)0, acc[j], /*reuse_a=*/false, /*reuse_b=*/false);
    }
  }

  const int rbase = khalf << 3;
  const int ncol0 = (ns << 6) + (lane & 15);
#pragma unroll
  for (int j = 0; j < 4; ++j) {
    const int n  = ncol0 + (j << 4);
    const float bv = bias ? bias[n] : 0.0f;
    float* op = Out + (size_t)((mt << 4) + rbase) * Ncols + n;
#pragma unroll
    for (int r = 0; r < 8; ++r)
      op[(size_t)r * Ncols] = acc[j][r] + bv;
  }
}

// ---------------- edge scatter: ACC[dst] += dinv[src]*w*dinv[dst] * XW[src] ----------------
// One wave32 per edge, 4 features per lane (coalesced float4 gather, native f32 atomics).
__global__ __launch_bounds__(256) void k_scatter(const int* __restrict__ src,
                                                 const int* __restrict__ dst,
                                                 const float* __restrict__ ew,
                                                 const float* __restrict__ dinv,
                                                 const float* __restrict__ XW,
                                                 float* __restrict__ ACC) {
  const int e = (blockIdx.x << 3) + (threadIdx.x >> 5);
  if (e >= NE) return;
  const int lane = threadIdx.x & 31;
  const int s = src[e], d = dst[e];
  const float nrm = dinv[s] * ew[e] * dinv[d];
  const float4 v = *(const float4*)(XW + (size_t)s * HID + (lane << 2));
  float* out = ACC + (size_t)d * HID + (lane << 2);
  __hip_atomic_fetch_add(out + 0, nrm * v.x, __ATOMIC_RELAXED, __HIP_MEMORY_SCOPE_AGENT);
  __hip_atomic_fetch_add(out + 1, nrm * v.y, __ATOMIC_RELAXED, __HIP_MEMORY_SCOPE_AGENT);
  __hip_atomic_fetch_add(out + 2, nrm * v.z, __ATOMIC_RELAXED, __HIP_MEMORY_SCOPE_AGENT);
  __hip_atomic_fetch_add(out + 3, nrm * v.w, __ATOMIC_RELAXED, __HIP_MEMORY_SCOPE_AGENT);
}

// ---------------- epilogue: h = [relu](ACC + dinv^2 * XW + b) -> bf16 for next GEMM --------
__global__ __launch_bounds__(256) void k_finish(const float* __restrict__ ACC,
                                                const float* __restrict__ XW,
                                                const float* __restrict__ dinv,
                                                const float* __restrict__ bias,
                                                __bf16* __restrict__ Abf, int do_relu) {
  const int idx = blockIdx.x * 256 + threadIdx.x;
  if (idx >= NN * HID) return;
  const int i = idx >> 7;
  const int f = idx & 127;
  const float di = dinv[i];
  float v = ACC[idx] + di * di * XW[idx] + bias[f];   // self-loop folded in analytically
  if (do_relu) v = fmaxf(v, 0.0f);
  Abf[idx] = f32_to_bf16(v);
}

extern "C" void kernel_launch(void* const* d_in, const int* in_sizes, int n_in,
                              void* d_out, int out_size, void* d_ws, size_t ws_size,
                              hipStream_t stream) {
  const float* x   = (const float*)d_in[0];
  const int*   ei  = (const int*)d_in[1];
  const float* ew  = (const float*)d_in[2];
  const float* W1  = (const float*)d_in[3];
  const float* b1  = (const float*)d_in[4];
  const float* W2  = (const float*)d_in[5];
  const float* b2  = (const float*)d_in[6];
  const float* Wmu = (const float*)d_in[7];
  const float* bmu = (const float*)d_in[8];
  const float* Wlv = (const float*)d_in[9];
  const float* blv = (const float*)d_in[10];
  const int* srcIdx = ei;
  const int* dstIdx = ei + NE;
  float* out = (float*)d_out;

  char* ws = (char*)d_ws;
  size_t off = 0;
  auto alloc = [&](size_t bytes) -> void* {
    void* p = ws + off;
    off += (bytes + 255) & ~(size_t)255;
    return p;
  };
  float*        deg  = (float*)alloc((size_t)NN * 4);
  float*        dinv = (float*)alloc((size_t)NN * 4);
  __bf16*       Abf  = (__bf16*)alloc((size_t)NN * HID * 2);
  float*        XW   = (float*)alloc((size_t)NN * HID * 4);
  float*        ACC  = (float*)alloc((size_t)NN * HID * 4);
  unsigned int* Wp1  = (unsigned int*)alloc((size_t)HID * HID * 2);
  unsigned int* Wp2  = (unsigned int*)alloc((size_t)HID * HID * 2);
  unsigned int* Wpmu = (unsigned int*)alloc((size_t)HID * LAT * 2);
  unsigned int* Wplv = (unsigned int*)alloc((size_t)HID * LAT * 2);

  const int nBlkN   = (NN + 255) / 256;        // 196
  const int nBlkNF  = (NN * HID) / 256;        // 25000
  const int nBlkAcc = (NN * HID / 4) / 256;    // 6250
  const int gemmBlk128 = ((NN / 16) * 2 + 3) / 4;  // 1563
  const int gemmBlk64  = ((NN / 16) * 1 + 3) / 4;  // 782

  // normalization
  k_fill<<<nBlkN, 256, 0, stream>>>(deg, 1.0f, NN);                 // self-loop weight
  k_deg_accum<<<NE / 256, 256, 0, stream>>>(dstIdx, ew, deg);
  k_dinv<<<nBlkN, 256, 0, stream>>>(deg, dinv, NN);

  // repack weights to WMMA B-fragment layout (bf16)
  k_pack_w<<<(HID * HID / 2 + 255) / 256, 256, 0, stream>>>(W1,  Wp1,  HID, HID);
  k_pack_w<<<(HID * HID / 2 + 255) / 256, 256, 0, stream>>>(W2,  Wp2,  HID, HID);
  k_pack_w<<<(HID * LAT / 2 + 255) / 256, 256, 0, stream>>>(Wmu, Wpmu, HID, LAT);
  k_pack_w<<<(HID * LAT / 2 + 255) / 256, 256, 0, stream>>>(Wlv, Wplv, HID, LAT);

  // layer 1
  k_cvt_bf16<<<nBlkNF, 256, 0, stream>>>(x, Abf, NN * HID);
  k_gemm_bf16<<<gemmBlk128, 128, 0, stream>>>(Abf, Wp1, nullptr, XW, NN, HID);
  k_zero4<<<nBlkAcc, 256, 0, stream>>>((float4*)ACC, NN * HID / 4);
  k_scatter<<<NE / 8, 256, 0, stream>>>(srcIdx, dstIdx, ew, dinv, XW, ACC);
  k_finish<<<nBlkNF, 256, 0, stream>>>(ACC, XW, dinv, b1, Abf, 1);

  // layer 2
  k_gemm_bf16<<<gemmBlk128, 128, 0, stream>>>(Abf, Wp2, nullptr, XW, NN, HID);
  k_zero4<<<nBlkAcc, 256, 0, stream>>>((float4*)ACC, NN * HID / 4);
  k_scatter<<<NE / 8, 256, 0, stream>>>(srcIdx, dstIdx, ew, dinv, XW, ACC);
  k_finish<<<nBlkNF, 256, 0, stream>>>(ACC, XW, dinv, b2, Abf, 0);

  // heads: mu, logvar straight into d_out (bias fused in GEMM epilogue)
  k_gemm_bf16<<<gemmBlk64, 128, 0, stream>>>(Abf, Wpmu, bmu, out, NN, LAT);
  k_gemm_bf16<<<gemmBlk64, 128, 0, stream>>>(Abf, Wplv, blv, out + (size_t)NN * LAT, NN, LAT);
}
